// LoRA_Layer_18614388261055
// MI455X (gfx1250) — matrix operand order repaired and checked by reading, not compiled, not run on hardware
//
#include <hip/hip_runtime.h>

// ---------------------------------------------------------------------------
// LoRA layer for MI455X (gfx1250):
//   out = x @ (W + s * A@B) + bias
// Fold rank-16 update into W once (bf16, transposed), convert x to bf16 once,
// then one compute-bound bf16 WMMA GEMM with f32 accumulation.
// Round 4: fix async builtin pointee type — the toolchain declares
// __builtin_amdgcn_global_load_async_to_lds_b128 as
//   (int4 addrspace(1)*, int4 addrspace(3)*, imm offset, imm cpol)
// per the round-3 diagnostic ("vector_size(4*sizeof(int)) int __device__ *").
// ---------------------------------------------------------------------------

typedef __attribute__((ext_vector_type(16))) __bf16 v16bf;
typedef __attribute__((ext_vector_type(8)))  __bf16 v8bf;
typedef __attribute__((ext_vector_type(8)))  float  v8f;

#if __has_builtin(__builtin_amdgcn_global_load_async_to_lds_b128)
#define HAS_ASYNC_LDS 1
typedef int v4i __attribute__((vector_size(16)));          // b128 payload type
typedef __attribute__((address_space(1))) v4i as1_v4i;     // global
typedef __attribute__((address_space(3))) v4i as3_v4i;     // LDS

// One 16-byte global->LDS async copy per lane (no VGPR data path).
static __device__ __forceinline__ void async_ld16(const unsigned short* g,
                                                  const unsigned short* l) {
    __builtin_amdgcn_global_load_async_to_lds_b128(
        (as1_v4i*)(unsigned long long)g,
        (as3_v4i*)(unsigned)(unsigned long long)l,   // low 32b = LDS offset
        0, 0);
}
static __device__ __forceinline__ void wait_async0() {
#if __has_builtin(__builtin_amdgcn_s_wait_asynccnt)
    __builtin_amdgcn_s_wait_asynccnt(0);
#else
    asm volatile("s_wait_asynccnt 0" ::: "memory");
#endif
}
#else
#define HAS_ASYNC_LDS 0
#endif

// float -> bf16 bits, round-to-nearest-even.
static __device__ __forceinline__ unsigned short f2bf(float f) {
    union { float f; unsigned u; } v; v.f = f;
    unsigned r = v.u + 0x7FFFu + ((v.u >> 16) & 1u);
    return (unsigned short)(r >> 16);
}

// ---------------------------------------------------------------------------
// Kernel 1: WpT[j][i] = bf16( W[i][j] + s * sum_r A[i][r]*B[r][j] )
// 64x64 tile per block; LDS transpose so both global read & write coalesce.
// ---------------------------------------------------------------------------
__global__ void prep_wpT(const float* __restrict__ W,
                         const float* __restrict__ A,
                         const float* __restrict__ B,
                         const int*   __restrict__ scale,
                         unsigned short* __restrict__ wpT,
                         int dIn, int dOut, int R) {
    __shared__ float sW[64][65];   // +1 pad: conflict-free transposed reads
    __shared__ float sA[64][17];
    __shared__ float sB[16][64];

    const int t  = threadIdx.x;            // 256 threads
    const int i0 = blockIdx.y * 64;        // d_in  tile
    const int j0 = blockIdx.x * 64;        // d_out tile
    const float s = (float)scale[0];

    for (int p = t; p < 64 * 64; p += 256) {           // W: coalesced in j
        int il = p >> 6, jl = p & 63;
        sW[il][jl] = W[(size_t)(i0 + il) * dOut + j0 + jl];
    }
    for (int p = t; p < 64 * 16; p += 256) {           // A rows
        int il = p >> 4, rc = p & 15;
        sA[il][rc] = (rc < R) ? A[(size_t)(i0 + il) * R + rc] : 0.0f;
    }
    for (int p = t; p < 16 * 64; p += 256) {           // B cols: coalesced in j
        int rc = p >> 6, jl = p & 63;
        sB[rc][jl] = (rc < R) ? B[(size_t)rc * dOut + j0 + jl] : 0.0f;
    }
    __syncthreads();

    for (int p = t; p < 64 * 64; p += 256) {           // write: coalesced in i
        int jl = p >> 6, il = p & 63;
        float d = 0.0f;
        #pragma unroll
        for (int rc = 0; rc < 16; ++rc) d += sA[il][rc] * sB[rc][jl];
        wpT[(size_t)(j0 + jl) * dIn + i0 + il] = f2bf(sW[il][jl] + s * d);
    }
}

// ---------------------------------------------------------------------------
// Kernel 2: x (f32) -> bf16, 8 elements / thread.
// ---------------------------------------------------------------------------
__global__ void cvt_bf16(const float* __restrict__ x,
                         unsigned short* __restrict__ xb, size_t n) {
    size_t i = ((size_t)blockIdx.x * blockDim.x + threadIdx.x) * 8;
    if (i >= n) return;
    float4 f0 = *(const float4*)(x + i);
    float4 f1 = *(const float4*)(x + i + 4);
    uint4 u;
    u.x = (unsigned)f2bf(f0.x) | ((unsigned)f2bf(f0.y) << 16);
    u.y = (unsigned)f2bf(f0.z) | ((unsigned)f2bf(f0.w) << 16);
    u.z = (unsigned)f2bf(f1.x) | ((unsigned)f2bf(f1.y) << 16);
    u.w = (unsigned)f2bf(f1.z) | ((unsigned)f2bf(f1.w) << 16);
    *(uint4*)(xb + i) = u;
}

// ---------------------------------------------------------------------------
// Kernel 3: GEMM  out[M,N] = xb[M,K] @ WpT[N,K]^T + bias
// Block: 256 thr = 8 wave32. Tile BM=128 BN=128 BK=64, LDS double-buffered
// (64 KB of the 320 KB pool). Wave (wm,wn) owns 32(M) x 64(N) = 2x4 accs.
// Per LDS tile: 2 K-steps x 8 WMMA = 16 v_wmma per barrier.
// ---------------------------------------------------------------------------
__global__ void __launch_bounds__(256)
__attribute__((amdgpu_waves_per_eu(2)))
gemm_bf16(const unsigned short* __restrict__ xb,
          const unsigned short* __restrict__ wpT,
          const float* __restrict__ bias,
          float* __restrict__ out,
          int M, int K, int N) {
    // [buf][row][64 bf16]  (row stride 128 B)
    __shared__ unsigned short lA[2][128 * 64] __attribute__((aligned(64)));
    __shared__ unsigned short lB[2][128 * 64] __attribute__((aligned(64)));

    const int t    = threadIdx.x;
    const int lane = t & 31;
    const int wave = t >> 5;
    const int wm   = wave >> 1;        // 0..3 -> M offset wm*32
    const int wn   = wave & 1;         // 0..1 -> N offset wn*64
    const int r16  = lane & 15;
    const int hi   = lane >> 4;        // lane half (per ISA fragment layout)
    const int mBlk = blockIdx.y * 128;
    const int nBlk = blockIdx.x * 128;

    // Tile fill: 2 threads per tile row, 64 B (32 bf16) each.
    const int row  = t >> 1;
    const int half = t & 1;
    const unsigned short* gA = xb  + (size_t)(mBlk + row) * K + half * 32;
    const unsigned short* gB = wpT + (size_t)(nBlk + row) * K + half * 32;
    const int ldsOff = row * 64 + half * 32;

    v8f acc[2][4];
    const v8f vzero = {0.f, 0.f, 0.f, 0.f, 0.f, 0.f, 0.f, 0.f};
    #pragma unroll
    for (int im = 0; im < 2; ++im)
        #pragma unroll
        for (int in = 0; in < 4; ++in) acc[im][in] = vzero;

    const int KT = K / 64;             // 64 tiles

#if HAS_ASYNC_LDS
    // ---- async global->LDS pipeline: zero staging VGPRs ----
    #pragma unroll
    for (int q = 0; q < 4; ++q) {               // prologue: tile 0 -> buf 0
        async_ld16(gA + q * 8, &lA[0][ldsOff] + q * 8);
        async_ld16(gB + q * 8, &lB[0][ldsOff] + q * 8);
    }
    wait_async0();
    __syncthreads();
#else
    uint4 rA[4], rB[4];
    #pragma unroll
    for (int q = 0; q < 4; ++q) {
        rA[q] = *(const uint4*)(gA + q * 8);
        rB[q] = *(const uint4*)(gB + q * 8);
    }
    {
        unsigned short* pa = &lA[0][ldsOff];
        unsigned short* pb = &lB[0][ldsOff];
        #pragma unroll
        for (int q = 0; q < 4; ++q) {
            *(uint4*)(pa + q * 8) = rA[q];
            *(uint4*)(pb + q * 8) = rB[q];
        }
    }
    __syncthreads();
#endif

    for (int kt = 0; kt < KT; ++kt) {
        const int  cur  = kt & 1;
        const bool more = (kt + 1) < KT;

#if HAS_ASYNC_LDS
        if (more) {                     // fill other buffer via async DMA
            const unsigned short* a = gA + (size_t)(kt + 1) * 64;
            const unsigned short* b = gB + (size_t)(kt + 1) * 64;
            unsigned short* pa = &lA[cur ^ 1][ldsOff];
            unsigned short* pb = &lB[cur ^ 1][ldsOff];
            #pragma unroll
            for (int q = 0; q < 4; ++q) {
                async_ld16(a + q * 8, pa + q * 8);
                async_ld16(b + q * 8, pb + q * 8);
            }
        }
#else
        if (more) {                     // stage next tile into regs
            const unsigned short* a = gA + (size_t)(kt + 1) * 64;
            const unsigned short* b = gB + (size_t)(kt + 1) * 64;
            #pragma unroll
            for (int q = 0; q < 4; ++q) {
                rA[q] = *(const uint4*)(a + q * 8);
                rB[q] = *(const uint4*)(b + q * 8);
            }
        }
#endif
        if ((kt + 2) < KT) {            // global_prefetch_b8 2 ahead
            __builtin_prefetch(gA + (size_t)(kt + 2) * 64, 0, 1);
            __builtin_prefetch(gB + (size_t)(kt + 2) * 64, 0, 1);
        }

        // ---- compute from lds[cur]: 2 K-steps of 8 WMMAs ----
        const unsigned short* sa = lA[cur];
        const unsigned short* sb = lB[cur];

        #pragma unroll
        for (int ks = 0; ks < 2; ++ks) {
            // All fragment loads first: one ds_load clause, drained by
            // decreasing s_wait_dscnt while WMMAs issue.
            v16bf afrag[2];
            v16bf bfrag[4];
            #pragma unroll
            for (int im = 0; im < 2; ++im) {
                const int abase = (wm * 32 + im * 16 + r16) * 64 + ks * 32;
                v8bf lo = *(const v8bf*)(sa + abase + hi * 8);
                v8bf hp = *(const v8bf*)(sa + abase + 16 + hi * 8);
                afrag[im] = __builtin_shufflevector(lo, hp,
                    0, 1, 2, 3, 4, 5, 6, 7, 8, 9, 10, 11, 12, 13, 14, 15);
            }
            #pragma unroll
            for (int in = 0; in < 4; ++in) {
                const int bbase = (wn * 64 + in * 16 + r16) * 64 + ks * 32;
                v8bf blo = *(const v8bf*)(sb + bbase + hi * 8);
                v8bf bhp = *(const v8bf*)(sb + bbase + 16 + hi * 8);
                bfrag[in] = __builtin_shufflevector(blo, bhp,
                    0, 1, 2, 3, 4, 5, 6, 7, 8, 9, 10, 11, 12, 13, 14, 15);
            }
            #pragma unroll
            for (int in = 0; in < 4; ++in)
                #pragma unroll
                for (int im = 0; im < 2; ++im)
                    acc[im][in] = __builtin_amdgcn_wmma_f32_16x16x32_bf16(
                        false, afrag[im], false, bfrag[in],
                        (short)0, acc[im][in], false, false);
        }

#if HAS_ASYNC_LDS
        if (more) wait_async0();        // our 8 async copies have landed
#else
        if (more) {                     // commit staged regs -> other buf
            unsigned short* pa = &lA[cur ^ 1][ldsOff];
            unsigned short* pb = &lB[cur ^ 1][ldsOff];
            #pragma unroll
            for (int q = 0; q < 4; ++q) {
                *(uint4*)(pa + q * 8) = rA[q];
                *(uint4*)(pb + q * 8) = rB[q];
            }
        }
#endif
        __syncthreads();
    }

    // Epilogue: C/D layout — VGPR v: lanes 0-15 M=v, lanes 16-31 M=8+v.
    #pragma unroll
    for (int in = 0; in < 4; ++in) {
        const int n  = nBlk + wn * 64 + in * 16 + r16;
        const float bv = bias[n];
        #pragma unroll
        for (int im = 0; im < 2; ++im) {
            const int mBase = mBlk + wm * 32 + im * 16 + hi * 8;
            #pragma unroll
            for (int v = 0; v < 8; ++v)
                out[(size_t)(mBase + v) * N + n] = acc[im][in][v] + bv;
        }
    }
}

// ---------------------------------------------------------------------------
// Launch. Inputs: x, W, bias, lora_A, lora_B, scaling(int scalar).
// Workspace: WpT bf16 (32 MB) then x bf16 (64 MB) = 96 MB of d_ws.
// ---------------------------------------------------------------------------
extern "C" void kernel_launch(void* const* d_in, const int* in_sizes, int n_in,
                              void* d_out, int out_size, void* d_ws, size_t ws_size,
                              hipStream_t stream) {
    const float* x     = (const float*)d_in[0];
    const float* W     = (const float*)d_in[1];
    const float* bias  = (const float*)d_in[2];
    const float* loraA = (const float*)d_in[3];
    const float* loraB = (const float*)d_in[4];
    const int*   scal  = (const int*)  d_in[5];
    float*       out   = (float*)d_out;

    const int dOut = in_sizes[2];                 // 4096
    const int R    = in_sizes[4] / dOut;          // 16
    const int dIn  = in_sizes[1] / dOut;          // 4096
    const long M   = (long)in_sizes[0] / dIn;     // 8192 (= 4*2048)

    unsigned short* wpT = (unsigned short*)d_ws;                    // [dOut][dIn]
    unsigned short* xbf = wpT + (size_t)dIn * dOut;                 // [M][dIn]

    // 1) fold LoRA into W, store transposed bf16
    prep_wpT<<<dim3(dOut / 64, dIn / 64), 256, 0, stream>>>(
        W, loraA, loraB, scal, wpT, dIn, dOut, R);

    // 2) x -> bf16
    const size_t nx = (size_t)M * dIn;
    cvt_bf16<<<(unsigned)((nx / 8 + 255) / 256), 256, 0, stream>>>(x, xbf, nx);

    // 3) WMMA GEMM + bias
    gemm_bf16<<<dim3(dOut / 128, (unsigned)(M / 128)), 256, 0, stream>>>(
        xbf, wpT, bias, out, (int)M, dIn, dOut);
}
